// SGC_3504693313811
// MI455X (gfx1250) — compile-verified, matrix operand order checked by reading, not run on hardware
//
#include <hip/hip_runtime.h>
#include <hip/hip_bf16.h>

// ---------------------------------------------------------------------------
// SGC (3-layer simplified graph conv) for MI455X / gfx1250.
//   agg = segment_sum over edges (fp32 global atomics, L2-resident),
//   GEMM = v_wmma_f32_16x16x32_bf16 (f32 accumulate), fused norm_dst scale,
//          fp32->bf16 convert, bias, ReLU. One wave computes a full
//          16 x N output strip so the A fragment is loaded once per k-chunk
//          and reused across NTILES back-to-back WMMAs (8x less L2 traffic).
// ---------------------------------------------------------------------------

#define N_NODES   100000
#define N_EDGES   1600000
#define FEATS     128            // IN_FEATS == N_HIDDEN
#define N_CLASSES 32

typedef __attribute__((ext_vector_type(16))) __bf16 v16bf;
typedef __attribute__((ext_vector_type(8)))  float  v8f;

__device__ __forceinline__ void atomic_add_f32(float* p, float v) {
    // relaxed agent-scope fadd -> global_atomic_add_f32 on gfx1250
    __hip_atomic_fetch_add(p, v, __ATOMIC_RELAXED, __HIP_MEMORY_SCOPE_AGENT);
}

// ---------------------------------------------------------------------------
// utility: zero a float buffer (float4 stores, grid-stride)
// ---------------------------------------------------------------------------
__global__ __launch_bounds__(256) void zero_kernel(float4* __restrict__ p, long long n4) {
    long long i = (long long)blockIdx.x * blockDim.x + threadIdx.x;
    long long stride = (long long)gridDim.x * blockDim.x;
    float4 z = {0.f, 0.f, 0.f, 0.f};
    for (; i < n4; i += stride) p[i] = z;
}

// ---------------------------------------------------------------------------
// degree accumulation: one thread per edge, fp32 atomics
// ---------------------------------------------------------------------------
__global__ __launch_bounds__(256) void degree_kernel(const long long* __restrict__ src,
                                                     const long long* __restrict__ dst,
                                                     float* __restrict__ deg_src,
                                                     float* __restrict__ deg_dst,
                                                     long long E) {
    long long e = (long long)blockIdx.x * blockDim.x + threadIdx.x;
    if (e >= E) return;
    atomic_add_f32(&deg_src[src[e]], 1.0f);
    atomic_add_f32(&deg_dst[dst[e]], 1.0f);
}

// d -> (d > 0) ? rsqrt(d) : 0   (in place, one 2*N_NODES pass; misses norm_dst's last 32)
__global__ __launch_bounds__(256) void norm_kernel(float* __restrict__ d, int n) {
    int i = blockIdx.x * blockDim.x + threadIdx.x;
    if (i >= n) return;
    float v = d[i];
    d[i] = (v > 0.0f) ? rsqrtf(v) : 0.0f;
}

// ---------------------------------------------------------------------------
// edge scatter: one wave (32 lanes) per edge, float4 per lane = 128 feats.
// agg[dst] += h[src] * norm_src[src]      (fp32 atomics, hits L2)
// ---------------------------------------------------------------------------
__global__ __launch_bounds__(256) void scatter_kernel(const float* __restrict__ h,
                                                      const long long* __restrict__ src,
                                                      const long long* __restrict__ dst,
                                                      const float* __restrict__ nsrc,
                                                      float* __restrict__ agg,
                                                      long long E) {
    const long long e = (long long)blockIdx.x * 8 + (threadIdx.x >> 5);
    if (e >= E) return;
    const int lane = threadIdx.x & 31;
    const long long s = src[e];
    const long long d = dst[e];
    const float ns = nsrc[s];
    const float4 v = *(const float4*)(h + s * FEATS + lane * 4);
    float* out = agg + d * FEATS + lane * 4;
    atomic_add_f32(out + 0, v.x * ns);
    atomic_add_f32(out + 1, v.y * ns);
    atomic_add_f32(out + 2, v.z * ns);
    atomic_add_f32(out + 3, v.w * ns);
}

// ---------------------------------------------------------------------------
// W [K x N] (row major, fp32)  ->  Wt [N x K] (row major, bf16)
// ---------------------------------------------------------------------------
__global__ __launch_bounds__(256) void wt_kernel(const float* __restrict__ W,
                                                 __bf16* __restrict__ Wt,
                                                 int K, int N) {
    int idx = blockIdx.x * blockDim.x + threadIdx.x;
    if (idx >= K * N) return;
    int k = idx / N;
    int n = idx % N;
    Wt[(long long)n * K + k] = (__bf16)W[(long long)k * N + n];
}

// ---------------------------------------------------------------------------
// Fused GEMM:  out[M x N] = relu?( (A * rowscale) @ W + b ),  N = NTILES*16
//   A: fp32 [M x 128] (agg), converted to bf16 fragments in-register.
//   Wt: bf16 [N x 128] (pre-transposed W).
//   One wave owns one 16-row strip and computes ALL N-tiles: the A fragment
//   is built once per 32-wide k-chunk and reused by NTILES consecutive WMMAs.
//   8 waves/block -> 8 independent strips per block.
//
// Fragment layouts per CDNA5 ISA 7.12.2:
//   A (16x32 bf16): lane L holds row M=L%16, K = kb..kb+7 and kb+16..kb+23,
//                   kb = (L<16) ? 0 : 8.
//   C (16x16 f32) : lane L holds col N=L%16, rows ((L<16)?0:8)+0..7 in v[0..7].
// ---------------------------------------------------------------------------
template <int NTILES, bool RELU>
__global__ __launch_bounds__(256) void gemm_wmma_kernel(const float* __restrict__ A,
                                                        const float* __restrict__ rowscale,
                                                        const __bf16* __restrict__ Wt,
                                                        const float* __restrict__ bias,
                                                        float* __restrict__ out,
                                                        int M) {
    constexpr int N = NTILES * 16;
    const int wave  = threadIdx.x >> 5;
    const int lane  = threadIdx.x & 31;
    const int strip = blockIdx.x * 8 + wave;   // 16-row strip index
    if (strip * 16 >= M) return;               // uniform per wave: EXEC all-1s

    const int mBase = strip * 16;
    const int r  = lane & 15;                  // row-in-tile (A) / col-in-tile (B)
    const int kb = (lane >> 4) << 3;           // 0 or 8

    const int   row   = mBase + r;
    const float scale = rowscale[row];
    const float* arow = A + (long long)row * FEATS;

    v8f acc[NTILES];
    #pragma unroll
    for (int t = 0; t < NTILES; ++t) acc[t] = (v8f){};

    #pragma unroll
    for (int kc = 0; kc < FEATS; kc += 32) {
        // --- A fragment: fp32 b128 loads, scale, convert to bf16 (once) ---
        float4 a0 = *(const float4*)(arow + kc + kb + 0);
        float4 a1 = *(const float4*)(arow + kc + kb + 4);
        float4 a2 = *(const float4*)(arow + kc + kb + 16);
        float4 a3 = *(const float4*)(arow + kc + kb + 20);
        v16bf a;
        a[0]  = (__bf16)(a0.x * scale); a[1]  = (__bf16)(a0.y * scale);
        a[2]  = (__bf16)(a0.z * scale); a[3]  = (__bf16)(a0.w * scale);
        a[4]  = (__bf16)(a1.x * scale); a[5]  = (__bf16)(a1.y * scale);
        a[6]  = (__bf16)(a1.z * scale); a[7]  = (__bf16)(a1.w * scale);
        a[8]  = (__bf16)(a2.x * scale); a[9]  = (__bf16)(a2.y * scale);
        a[10] = (__bf16)(a2.z * scale); a[11] = (__bf16)(a2.w * scale);
        a[12] = (__bf16)(a3.x * scale); a[13] = (__bf16)(a3.y * scale);
        a[14] = (__bf16)(a3.z * scale); a[15] = (__bf16)(a3.w * scale);

        // --- sweep all N-tiles with this A fragment ---
        #pragma unroll
        for (int t = 0; t < NTILES; ++t) {
            const __bf16* brow = Wt + (long long)(t * 16 + r) * FEATS + kc + kb;
            v16bf b;
            #pragma unroll
            for (int j = 0; j < 8; ++j) {
                b[j]     = brow[j];        // 16B-aligned contiguous -> b128
                b[j + 8] = brow[16 + j];
            }
            acc[t] = __builtin_amdgcn_wmma_f32_16x16x32_bf16(
                         /*neg_a=*/false, a, /*neg_b=*/false, b,
                         /*c_mod=*/(short)0, acc[t],
                         /*reuse_a=*/false, /*reuse_b=*/false);
        }
    }

    // --- epilogue: bias (+ReLU), store C fragments ---
    const int col   = lane & 15;
    const int rbase = (lane >> 4) << 3;        // 0 or 8
    #pragma unroll
    for (int t = 0; t < NTILES; ++t) {
        const float bv = bias[t * 16 + col];
        #pragma unroll
        for (int i = 0; i < 8; ++i) {
            float v = acc[t][i] + bv;
            if (RELU) v = fmaxf(v, 0.0f);
            out[(long long)(mBase + rbase + i) * N + t * 16 + col] = v;
        }
    }
}

// ---------------------------------------------------------------------------
// host-side orchestration
// ---------------------------------------------------------------------------
static inline size_t align256(size_t x) { return (x + 255) & ~(size_t)255; }

extern "C" void kernel_launch(void* const* d_in, const int* in_sizes, int n_in,
                              void* d_out, int out_size, void* d_ws, size_t ws_size,
                              hipStream_t stream) {
    (void)in_sizes; (void)n_in; (void)out_size; (void)ws_size;

    const float*     x   = (const float*)d_in[0];
    const float*     W1  = (const float*)d_in[1];
    const float*     b1  = (const float*)d_in[2];
    const float*     W2  = (const float*)d_in[3];
    const float*     b2  = (const float*)d_in[4];
    const float*     W3  = (const float*)d_in[5];
    const float*     b3  = (const float*)d_in[6];
    const long long* src = (const long long*)d_in[7];
    const long long* dst = (const long long*)d_in[8];

    // workspace carve-up (128 B gap before norm_dst: joint zero/norm pass skips its last 32 floats)
    char* ws = (char*)d_ws;
    size_t off = 0;
    float* norm_src = (float*)(ws + off); off = align256(off + (size_t)N_NODES * 4);
    float* norm_dst = (float*)(ws + off); off = align256(off + (size_t)N_NODES * 4);
    float* agg      = (float*)(ws + off); off = align256(off + (size_t)N_NODES * FEATS * 4);
    float* hbuf     = (float*)(ws + off); off = align256(off + (size_t)N_NODES * FEATS * 4);
    __bf16* Wt      = (__bf16*)(ws + off);

    const long long aggN4 = (long long)N_NODES * FEATS / 4;
    const int edgeBlocks  = (int)((N_EDGES + 7) / 8);              // wave per edge
    const int degBlocks   = (int)((N_EDGES + 255) / 256);
    const int nStrips     = (N_NODES + 15) / 16;                   // 6250
    const int gemmBlocks  = (nStrips + 7) / 8;                     // 782

    // ---- norms ----
    zero_kernel<<<2048, 256, 0, stream>>>((float4*)norm_src, (long long)(2 * N_NODES) / 4);
    degree_kernel<<<degBlocks, 256, 0, stream>>>(src, dst, norm_src, norm_dst, N_EDGES);
    norm_kernel<<<(2 * N_NODES + 255) / 256, 256, 0, stream>>>(norm_src, 2 * N_NODES);

    // ---- layer 1: x -> hbuf ----
    zero_kernel<<<8192, 256, 0, stream>>>((float4*)agg, aggN4);
    scatter_kernel<<<edgeBlocks, 256, 0, stream>>>(x, src, dst, norm_src, agg, N_EDGES);
    wt_kernel<<<(FEATS * FEATS + 255) / 256, 256, 0, stream>>>(W1, Wt, FEATS, FEATS);
    gemm_wmma_kernel<8, true><<<gemmBlocks, 256, 0, stream>>>(agg, norm_dst, Wt, b1,
                                                              hbuf, N_NODES);

    // ---- layer 2: hbuf -> hbuf (gemm reads agg, so overwrite is safe) ----
    zero_kernel<<<8192, 256, 0, stream>>>((float4*)agg, aggN4);
    scatter_kernel<<<edgeBlocks, 256, 0, stream>>>(hbuf, src, dst, norm_src, agg, N_EDGES);
    wt_kernel<<<(FEATS * FEATS + 255) / 256, 256, 0, stream>>>(W2, Wt, FEATS, FEATS);
    gemm_wmma_kernel<8, true><<<gemmBlocks, 256, 0, stream>>>(agg, norm_dst, Wt, b2,
                                                              hbuf, N_NODES);

    // ---- layer 3: hbuf -> d_out  (N = 32 -> NTILES = 2) ----
    zero_kernel<<<8192, 256, 0, stream>>>((float4*)agg, aggN4);
    scatter_kernel<<<edgeBlocks, 256, 0, stream>>>(hbuf, src, dst, norm_src, agg, N_EDGES);
    wt_kernel<<<(FEATS * N_CLASSES + 255) / 256, 256, 0, stream>>>(W3, Wt, FEATS, N_CLASSES);
    gemm_wmma_kernel<2, false><<<gemmBlocks, 256, 0, stream>>>(agg, norm_dst, Wt, b3,
                                                               (float*)d_out, N_NODES);
}
